// RNNTransducer_56487409877034
// MI455X (gfx1250) — matrix operand order repaired and checked by reading, not compiled, not run on hardware
//
#include <hip/hip_runtime.h>

// ---------------- WMMA types / helpers (CDNA5 gfx1250, wave32) ----------------
typedef __attribute__((ext_vector_type(16))) _Float16 v16h;
typedef __attribute__((ext_vector_type(8)))  float    v8f;

#define NBLK 16      // workgroups per persistent kernel (one cluster's worth)
#define BATCH 8
#define TT 256
#define UU 64
#define VV 8192
#define PP 640
#define HENC 1024

__device__ __forceinline__ v8f wmma16(v16h a, v16h b, v8f c) {
  // emits v_wmma_f32_16x16x32_f16
  return __builtin_amdgcn_wmma_f32_16x16x32_f16(
      /*neg_a=*/false, a, /*neg_b=*/false, b,
      /*c_mod=*/(short)0, c, /*reuse_a=*/false, /*reuse_b=*/false);
}

// A fragment: 16x32 f16 tile (rows = M), row-major in LDS with stride lda.
// ISA 7.12.2 16-bit A layout: lane m=L&15, kh=L>>4; VGPR v<4: K=kh*8+2v ; v>=4: K=16+kh*8+2(v-4)
__device__ __forceinline__ v16h load_A_frag(const _Float16* As, int lda, int kbase) {
  int lane = threadIdx.x & 31;
  int m  = lane & 15;
  int kh = lane >> 4;
  const _Float16* row = As + m * lda + kbase;
  v16h a;
#pragma unroll
  for (int v = 0; v < 8; ++v) {
    int grp = v >> 2, w = v & 3;
    int k = grp * 16 + kh * 8 + w * 2;
    a[2 * v]     = row[k];
    a[2 * v + 1] = row[k + 1];
  }
  return a;
}

// B fragment: prepacked fragment-major; each lane loads 16 contiguous f16 (32B).
__device__ __forceinline__ v16h load_B_frag(const _Float16* Bp) {
  int lane = threadIdx.x & 31;
  return *((const v16h*)Bp + lane);
}

__device__ __forceinline__ float sigf(float x) { return 1.f / (1.f + __expf(-x)); }

// 16B global -> LDS async copy (GLOBAL_LOAD_ASYNC_TO_LDS_B128, ASYNCcnt-tracked)
__device__ __forceinline__ void async_copy16(void* lds_dst, const void* gsrc) {
  unsigned lo = (unsigned)(size_t)lds_dst;               // LDS byte offset (addr[31:0])
  unsigned long long ga = (unsigned long long)(size_t)gsrc;
  asm volatile("global_load_async_to_lds_b128 %0, %1, off"
               :: "v"(lo), "v"(ga) : "memory");
}
__device__ __forceinline__ void async_wait() {
  asm volatile("s_wait_asynccnt 0x0" ::: "memory");
}

// Monotonic grid-wide barrier across NBLK blocks + CDNA5 cluster barrier
// (s_barrier_signal/-wait -3; NOP when not dispatched as a cluster).
__device__ __forceinline__ void grid_barrier(unsigned* cnt) {
  __syncthreads();
  if (threadIdx.x == 0) {
    __threadfence();
    unsigned arrive = atomicAdd(cnt, 1u) + 1u;
    unsigned target = ((arrive + NBLK - 1u) / NBLK) * NBLK;
    volatile unsigned* v = (volatile unsigned*)cnt;
    while (*v < target) __builtin_amdgcn_s_sleep(2);
    __threadfence();
  }
  __builtin_amdgcn_s_cluster_barrier();
  __syncthreads();
}

// ---------------- weight prepack: f32 [K][N] -> f16 WMMA-B fragment-major ----------------
// layout: [(tn*nck + ck)][lane(32)][16 f16]; B-frag: k = ck*32 + (s>>3)*16 + (lane>>4)*8 + (s&7)
__global__ void pack_b_kernel(const float* __restrict__ W, int K, int N, int nck,
                              _Float16* __restrict__ out) {
  long total = (long)(N / 16) * nck * 512;
  for (long p = (long)blockIdx.x * blockDim.x + threadIdx.x; p < total;
       p += (long)gridDim.x * blockDim.x) {
    int  s  = (int)(p & 15);
    int  l  = (int)((p >> 4) & 31);
    long cg = p >> 9;
    int  ck = (int)(cg % nck);
    int  tn = (int)(cg / nck);
    int  n  = tn * 16 + (l & 15);
    int  k  = ck * 32 + (s >> 3) * 16 + (l >> 4) * 8 + (s & 7);
    float val = (k < K) ? W[(long)k * N + n] : 0.f;
    out[p] = (_Float16)val;
  }
}

// elementwise f32 -> f16
__global__ void cvt_f16_kernel(const float* __restrict__ src, _Float16* __restrict__ dst, long n) {
  for (long i = (long)blockIdx.x * blockDim.x + threadIdx.x; i < n;
       i += (long)gridDim.x * blockDim.x)
    dst[i] = (_Float16)src[i];
}

// ---------------- persistent LSTMP encoder layer ----------------
// 16 WGs; wg owns 64 hidden units. Per step: async-stage A=[x_t|h] (f16) -> gate WMMA ->
// cell elementwise -> grid sync -> async-stage m -> projection WMMA -> grid sync.
// Unified LDS row stride L = max(Kpad,1024); rows 8..15 zeroed once, never rewritten.
__global__ __launch_bounds__(256)
void enc_layer_kernel(const _Float16* __restrict__ xin16, int din,
                      const _Float16* __restrict__ Wg,   // packed [256 tiles][nck][512]
                      const float* __restrict__ bg,      // [4096]
                      const _Float16* __restrict__ Wp,   // packed [40 tiles][32][512]
                      float* __restrict__ ysf32,         // [B,T,P]
                      _Float16* __restrict__ ysf16,      // [B,T,P]
                      float* cstate, _Float16* h16, _Float16* mbuf16,
                      unsigned* barcnt) {
  __shared__ __align__(16) _Float16 As[16 * 1280];
  __shared__ float zbuf[4][BATCH][64];

  const int tid = threadIdx.x, wg = blockIdx.x;
  const int wv = tid >> 5, lane = tid & 31;
  const int K = din + PP;
  const int nck = (K + 31) >> 5;
  const int Kpad = nck << 5;
  const int L = (Kpad > HENC) ? Kpad : HENC;   // unified LDS row stride

  for (int i = tid; i < BATCH * HENC; i += 256) cstate[i] = 0.f;  // identical writes across wgs
  for (int i = tid; i < BATCH * PP; i += 256)   h16[i] = (_Float16)0.f;
  for (int i = tid; i < 16 * 1280; i += 256)    As[i] = (_Float16)0.f;
  grid_barrier(barcnt);

  const int cpr = K >> 3;     // 16B chunks per A row
  const int xc  = din >> 3;   // chunks in the x part
  const int padw = Kpad - K;  // pad cols (16 for layer0, 0 for layer1)

  for (int t = 0; t < TT; ++t) {
    // (1) async-stage A = [x_t | h_prev] f16 rows 0..7 (rows 8..15 stay zero)
    for (int c = tid; c < 8 * cpr; c += 256) {
      int r = c / cpr, j = c - r * cpr;
      if (j < xc)
        async_copy16(&As[r * L + j * 8], &xin16[((long)r * TT + t) * din + j * 8]);
      else
        async_copy16(&As[r * L + din + (j - xc) * 8], &h16[r * PP + (j - xc) * 8]);
    }
    if (padw)  // re-zero pad cols clobbered by projection staging
      for (int i = tid; i < 8 * padw; i += 256) {
        int r = i / padw, cc = K + (i - r * padw);
        As[r * L + cc] = (_Float16)0.f;
      }
    async_wait();
    __syncthreads();

    // (2) gate GEMM: 16 N-tiles per wg, 2 per wave
#pragma unroll
    for (int q = 0; q < 2; ++q) {
      int tl = wv * 2 + q;
      int gate = tl >> 2, hu16 = tl & 3;
      int tn = gate * 64 + wg * 4 + hu16;
      const _Float16* bp = Wg + (long)tn * nck * 512;
      v8f acc = {};
      for (int ck = 0; ck < nck; ++ck) {
        __builtin_prefetch(bp + (long)(ck + 1) * 512 + lane * 16, 0, 1);
        acc = wmma16(load_A_frag(As, L, ck << 5), load_B_frag(bp + (long)ck * 512), acc);
      }
      int n = lane & 15, kh = lane >> 4;
      int colbase = gate * HENC + wg * 64 + hu16 * 16;
#pragma unroll
      for (int r = 0; r < 8; ++r) {
        int m = r + 8 * kh;
        if (m < BATCH) zbuf[gate][m][hu16 * 16 + n] = acc[r] + bg[colbase + n];
      }
    }
    __syncthreads();

    // (3) LSTM cell elementwise on this wg's 64 hidden units -> mbuf16 (f16)
    for (int e = tid; e < BATCH * 64; e += 256) {
      int b = e >> 6, hu = e & 63;
      int gh = wg * 64 + hu;
      float iv = zbuf[0][b][hu], jv = zbuf[1][b][hu];
      float fv = zbuf[2][b][hu], ov = zbuf[3][b][hu];
      float c = sigf(fv + 1.f) * cstate[b * HENC + gh] + sigf(iv) * tanhf(jv);
      cstate[b * HENC + gh] = c;
      mbuf16[b * HENC + gh] = (_Float16)(sigf(ov) * tanhf(c));
    }
    grid_barrier(barcnt);

    // (4) projection h = m @ Wp : async-stage m rows 0..7 (stride L), 40 tiles over waves
    for (int c = tid; c < 8 * 128; c += 256) {
      int r = c >> 7, j = c & 127;
      async_copy16(&As[r * L + j * 8], &mbuf16[r * HENC + j * 8]);
    }
    async_wait();
    __syncthreads();

    int gwv = wv * NBLK + wg;
    if (gwv < 40) {
      const _Float16* bp = Wp + (long)gwv * 32 * 512;
      v8f acc = {};
      for (int ck = 0; ck < 32; ++ck)
        acc = wmma16(load_A_frag(As, L, ck << 5), load_B_frag(bp + (long)ck * 512), acc);
      int n = lane & 15, kh = lane >> 4;
      int col = gwv * 16 + n;
#pragma unroll
      for (int r = 0; r < 8; ++r) {
        int m = r + 8 * kh;
        if (m < BATCH) {
          float h = acc[r];
          h16[m * PP + col] = (_Float16)h;
          ysf32[((long)m * TT + t) * PP + col] = h;
          ysf16[((long)m * TT + t) * PP + col] = (_Float16)h;
        }
      }
    }
    grid_barrier(barcnt);
  }
}

// ---------------- persistent greedy decoder + joint ----------------
__global__ __launch_bounds__(256)
void dec_joint_kernel(const float* __restrict__ enc_out,  // [B,T,P] f32
                      const float* __restrict__ decW0,    // [641*4096], row 0 used
                      const float* __restrict__ decb0,    // [4096]
                      const _Float16* __restrict__ WpD,   // packed [40][32][512]
                      const _Float16* __restrict__ Wj,    // packed [512][20][512]
                      const float* __restrict__ jb,       // [8192]
                      float* __restrict__ out,            // hj [B,320,V] then tokens [B,U]
                      _Float16* mdbuf16, float* gbuf,
                      float* argval, int* argidx, int* st,
                      unsigned* barcnt) {
  __shared__ __align__(16) _Float16 As[16 * 1024];
  __shared__ float sval[8][BATCH];
  __shared__ int   sidx[8][BATCH];

  const int tid = threadIdx.x, wg = blockIdx.x;
  const int wv = tid >> 5, lane = tid & 31;
  int* tcur = st; int* ucur = st + 8; int* ylast = st + 16; int* done = st + 24;

  for (int i = tid; i < 16 * 1024; i += 256) As[i] = (_Float16)0.f;
  if (wg == 0) {
    for (int i = tid; i < 32; i += 256) st[i] = 0;
    float* tok = out + (long)BATCH * 320 * VV;
    for (int i = tid; i < BATCH * UU; i += 256) tok[i] = 0.f;
  }
  grid_barrier(barcnt);

  for (int s = 0; s < 320; ++s) {
    // (a) decoder cell (c=h=0 -> only dec_W0 row 0 matters) -> mdec[8][1024] f16
    for (int e = wg * 256 + tid; e < BATCH * HENC; e += NBLK * 256) {
      int b = e >> 10, hu = e & 1023;
      float y = (float)ylast[b];
      float iv = y * decW0[hu]            + decb0[hu];
      float jv = y * decW0[HENC + hu]     + decb0[HENC + hu];
      float ov = y * decW0[3 * HENC + hu] + decb0[3 * HENC + hu];
      float c = sigf(iv) * tanhf(jv);
      mdbuf16[e] = (_Float16)(sigf(ov) * tanhf(c));
    }
    grid_barrier(barcnt);

    // (b) g = mdec @ dec_Wp0 -> gbuf[8][640] : async-stage mdec rows 0..7
    for (int c = tid; c < 8 * 128; c += 256) {
      int r = c >> 7, j = c & 127;
      async_copy16(&As[r * HENC + j * 8], &mdbuf16[r * HENC + j * 8]);
    }
    async_wait();
    __syncthreads();
    int gwv = wv * NBLK + wg;
    if (gwv < 40) {
      const _Float16* bp = WpD + (long)gwv * 32 * 512;
      v8f acc = {};
      for (int ck = 0; ck < 32; ++ck)
        acc = wmma16(load_A_frag(As, HENC, ck << 5), load_B_frag(bp + (long)ck * 512), acc);
      int n = lane & 15, kh = lane >> 4;
#pragma unroll
      for (int r = 0; r < 8; ++r) {
        int m = r + 8 * kh;
        if (m < BATCH) gbuf[m * PP + gwv * 16 + n] = acc[r];
      }
    }
    grid_barrier(barcnt);

    // (c) X = enc_out[b][min(t,T-1)] + g ; hj = tanh(X @ joint_W + jb); per-b argmax
    for (int i = tid; i < 16 * PP; i += 256) {
      int row = i / PP, col = i - row * PP;
      float v = 0.f;
      if (row < BATCH) {
        int tb = tcur[row]; if (tb > TT - 1) tb = TT - 1;
        v = enc_out[((long)row * TT + tb) * PP + col] + gbuf[row * PP + col];
      }
      As[row * PP + col] = (_Float16)v;
    }
    __syncthreads();

    float bestv[8]; int besti[8];
#pragma unroll
    for (int r = 0; r < 8; ++r) { bestv[r] = -3.4e38f; besti[r] = 0; }

    for (int j = 0; j < 4; ++j) {
      int tn = wg * 32 + wv * 4 + j;          // 512 tiles over N=8192
      const _Float16* bp = Wj + (long)tn * 20 * 512;
      v8f acc = {};
      for (int ck = 0; ck < 20; ++ck) {
        __builtin_prefetch(bp + (long)(ck + 1) * 512 + lane * 16, 0, 1);
        acc = wmma16(load_A_frag(As, PP, ck << 5), load_B_frag(bp + (long)ck * 512), acc);
      }
      int n = lane & 15, kh = lane >> 4;
      int col = tn * 16 + n;
      float bias = jb[col];
#pragma unroll
      for (int r = 0; r < 8; ++r) {
        float hv = tanhf(acc[r] + bias);
        int m = r + 8 * kh;
        if (m < BATCH) {
          float sc = done[m] ? 0.f : 1.f;     // hs = where(active, hj, 0)
          out[((long)m * 320 + s) * VV + col] = sc * hv;
        }
        // lanes 0-15 hold valid rows b=r; reduce (max,first-idx) across the wave
        float v = (kh == 0) ? hv : -3.4e38f;
        int ix = col;
#pragma unroll
        for (int mk = 16; mk >= 1; mk >>= 1) {
          float ov2 = __shfl_xor(v, mk, 32);
          int   oi2 = __shfl_xor(ix, mk, 32);
          if (ov2 > v || (ov2 == v && oi2 < ix)) { v = ov2; ix = oi2; }
        }
        if (v > bestv[r] || (v == bestv[r] && ix < besti[r])) { bestv[r] = v; besti[r] = ix; }
      }
    }
    if (lane == 0) {
#pragma unroll
      for (int r = 0; r < 8; ++r) { sval[wv][r] = bestv[r]; sidx[wv][r] = besti[r]; }
    }
    __syncthreads();
    if (tid < BATCH) {
      float v = sval[0][tid]; int ix = sidx[0][tid];
      for (int w2 = 1; w2 < 8; ++w2) {
        float ov2 = sval[w2][tid]; int oi2 = sidx[w2][tid];
        if (ov2 > v || (ov2 == v && oi2 < ix)) { v = ov2; ix = oi2; }
      }
      argval[wg * BATCH + tid] = v; argidx[wg * BATCH + tid] = ix;
    }
    grid_barrier(barcnt);

    // (d) global argmax + greedy state update (wg0)
    if (wg == 0 && tid < BATCH) {
      float v = argval[tid]; int am = argidx[tid];
      for (int w2 = 1; w2 < NBLK; ++w2) {
        float ov2 = argval[w2 * BATCH + tid]; int oi2 = argidx[w2 * BATCH + tid];
        if (ov2 > v || (ov2 == v && oi2 < am)) { v = ov2; am = oi2; }
      }
      int t = tcur[tid], u = ucur[tid], yl = ylast[tid], dn = done[tid];
      int active = !dn;
      int blank  = (am == 0);
      int emit   = active && !blank;
      if (emit && u < UU) out[(long)BATCH * 320 * VV + tid * UU + u] = (float)am;
      int t2 = (active && blank) ? t + 1 : t;
      int u2 = emit ? u + 1 : u;
      tcur[tid]  = t2;
      ucur[tid]  = u2;
      ylast[tid] = emit ? am : yl;
      done[tid]  = dn | (t2 >= TT) | (u2 >= UU);
    }
    grid_barrier(barcnt);
  }
}

// ---------------- launch ----------------
extern "C" void kernel_launch(void* const* d_in, const int* in_sizes, int n_in,
                              void* d_out, int out_size, void* d_ws, size_t ws_size,
                              hipStream_t stream) {
  (void)in_sizes; (void)n_in; (void)out_size; (void)ws_size;
  const float* x       = (const float*)d_in[0];
  const float* enc_W0  = (const float*)d_in[1];
  const float* enc_b0  = (const float*)d_in[2];
  const float* enc_Wp0 = (const float*)d_in[3];
  const float* enc_W1  = (const float*)d_in[4];
  const float* enc_b1  = (const float*)d_in[5];
  const float* enc_Wp1 = (const float*)d_in[6];
  const float* dec_W0  = (const float*)d_in[7];
  const float* dec_b0  = (const float*)d_in[8];
  const float* dec_Wp0 = (const float*)d_in[9];
  const float* joint_W = (const float*)d_in[10];
  const float* joint_b = (const float*)d_in[11];
  float* out = (float*)d_out;

  char* ws = (char*)d_ws;
  size_t off = 0;
  auto alloc = [&](size_t bytes) -> void* {
    void* p = ws + off;
    off = (off + bytes + 255) & ~(size_t)255;
    return p;
  };
  unsigned* barcnt = (unsigned*)alloc(256);
  int*      st     = (int*)alloc(32 * sizeof(int));
  float*    argval = (float*)alloc(NBLK * BATCH * sizeof(float));
  int*      argidx = (int*)alloc(NBLK * BATCH * sizeof(int));
  _Float16* pkW0   = (_Float16*)alloc((size_t)256 * 23 * 512 * 2);
  _Float16* pkW1   = (_Float16*)alloc((size_t)256 * 40 * 512 * 2);
  _Float16* pkWp0  = (_Float16*)alloc((size_t)40 * 32 * 512 * 2);
  _Float16* pkWp1  = (_Float16*)alloc((size_t)40 * 32 * 512 * 2);
  _Float16* pkWpD  = (_Float16*)alloc((size_t)40 * 32 * 512 * 2);
  _Float16* pkWj   = (_Float16*)alloc((size_t)512 * 20 * 512 * 2);
  _Float16* xf16   = (_Float16*)alloc((size_t)BATCH * TT * 80 * 2);
  _Float16* ys0f16 = (_Float16*)alloc((size_t)BATCH * TT * PP * 2);
  _Float16* ys1f16 = (_Float16*)alloc((size_t)BATCH * TT * PP * 2);
  float* ys0f32  = (float*)alloc((size_t)BATCH * TT * PP * 4);
  float* encout  = (float*)alloc((size_t)BATCH * TT * PP * 4);
  float* cstate  = (float*)alloc((size_t)BATCH * HENC * 4);
  _Float16* h16    = (_Float16*)alloc((size_t)BATCH * PP * 2);
  _Float16* mbuf16 = (_Float16*)alloc((size_t)BATCH * HENC * 2);
  _Float16* mdbuf16= (_Float16*)alloc((size_t)BATCH * HENC * 2);
  float* gbuf    = (float*)alloc((size_t)BATCH * PP * 4);

  hipMemsetAsync(barcnt, 0, 256, stream);

  // one-time (per launch) weight prepack into WMMA-B fragment-major f16
  pack_b_kernel<<<256, 256, 0, stream>>>(enc_W0, 720, 4096, 23, pkW0);
  pack_b_kernel<<<256, 256, 0, stream>>>(enc_W1, 1280, 4096, 40, pkW1);
  pack_b_kernel<<<64, 256, 0, stream>>>(enc_Wp0, 1024, 640, 32, pkWp0);
  pack_b_kernel<<<64, 256, 0, stream>>>(enc_Wp1, 1024, 640, 32, pkWp1);
  pack_b_kernel<<<64, 256, 0, stream>>>(dec_Wp0, 1024, 640, 32, pkWpD);
  pack_b_kernel<<<256, 256, 0, stream>>>(joint_W, 640, 8192, 20, pkWj);
  cvt_f16_kernel<<<64, 256, 0, stream>>>(x, xf16, (long)BATCH * TT * 80);

  enc_layer_kernel<<<NBLK, 256, 0, stream>>>(xf16, 80, pkW0, enc_b0, pkWp0,
                                             ys0f32, ys0f16, cstate, h16, mbuf16, barcnt);
  enc_layer_kernel<<<NBLK, 256, 0, stream>>>(ys0f16, 640, pkW1, enc_b1, pkWp1,
                                             encout, ys1f16, cstate, h16, mbuf16, barcnt);
  dec_joint_kernel<<<NBLK, 256, 0, stream>>>(encout, dec_W0, dec_b0, pkWpD, pkWj,
                                             joint_b, out, mdbuf16, gbuf,
                                             argval, argidx, st, barcnt);
}
